// RSSM_46505905881504
// MI455X (gfx1250) — matrix operand order repaired and checked by reading, not compiled, not run on hardware
//
#include <hip/hip_runtime.h>
#include <hip/hip_bf16.h>

typedef __attribute__((ext_vector_type(16))) __bf16 v16bf;
typedef __attribute__((ext_vector_type(8)))  float  v8f;

#define S_   30
#define D_   200
#define H_   200
#define A_   6
#define E_   1024
#define B_   256
#define T_   64
#define OUTW 1160

// LDS bf16 pool offsets (elements). Lifetime overlay:
//  A_small [16][64]   live P0..P1
//  XH      [16][416]  live P0..P2   (GRU input concat [x|deter])
//  PARTS   [16][600]  live P2..P4   (pre-LN GRU parts)
//  DB      [16][224]  live P4..G3   (new deter, bf16)   -- overlays A_small/XH
//  XO      [16][224]  live G2..G3   (elu(imgo))         -- overlays XH
//  XO2     [16][224]  live G2..G3   (elu(obso))         -- overlays PARTS
#define OFF_A     0
#define OFF_XH    1024
#define OFF_PARTS 7680
#define OFF_DB    0
#define OFF_XO    3584
#define OFF_XO2   7680
#define SBF_SIZE  17280

struct RssmParams {
  const float* embed;
  const float* action;
  const float *inp_b, *inpf_b;
  const float *gru_b, *gru_ls, *gru_lb;
  const float *gruf_b, *gruf_ls, *gruf_lb;
  const float *imgo_b, *imgof_b;
  const float *obso_b, *obsof_b;
  const float *ims_b, *imsf_b, *obs_b, *obsf_b;
  const __bf16 *inp_wt, *inpf_wt, *gru_wt, *gruf_wt;
  const __bf16 *imgo_wt, *imgof_wt, *obso_wt, *obsof_wt;
  const __bf16 *ims_wt, *imsf_wt, *obs_wt, *obsf_wt;
  float* out;
};

// ---- WMMA fragment loaders (wave32, 16x16x32 bf16) ----
// A 16x32: lanes 0-15 row M=lane, K 0-7 then 16-23; lanes 16-31 same rows, K +8.
__device__ inline v16bf ld_a_lds(const __bf16* a, int stride, int k0, int lane) {
  int r = lane & 15, hi = lane >> 4;
  const __bf16* base = a + r * stride + k0 + hi * 8;
  union { v16bf v; uint4 q[2]; } f;
  f.q[0] = *(const uint4*)(const void*)(base);        // K +0..7
  f.q[1] = *(const uint4*)(const void*)(base + 16);   // K +16..23
  return f.v;
}

// B 32x16 from transposed weight WT[dout_pad][din_pad]:
// lanes 0-15 col N=lane, K 0-7 then 16-23; lanes 16-31 same cols, K +8.
__device__ inline v16bf ld_b_gl(const __bf16* wt, int dk, int n0, int k0, int lane) {
  int n = n0 + (lane & 15), hi = lane >> 4;
  const __bf16* base = wt + (size_t)n * dk + k0 + hi * 8;
  union { v16bf v; uint4 q[2]; } f;
  f.q[0] = *(const uint4*)(const void*)(base);        // K +0..7
  f.q[1] = *(const uint4*)(const void*)(base + 16);   // K +16..23
  return f.v;
}

// A fragment straight from global f32 (embed), converting to bf16.
__device__ inline v16bf ld_a_glf32(const float* src, long long rstride, int k0, int lane) {
  int r = lane & 15, hi = lane >> 4;
  const float* base = src + (long long)r * rstride + k0 + hi * 8;
  union { v16bf v; __bf16 h[16]; } f;
#pragma unroll
  for (int u = 0; u < 4; ++u) { f.h[2*u]   = (__bf16)base[2*u];      f.h[2*u+1] = (__bf16)base[2*u+1]; }
#pragma unroll
  for (int u = 0; u < 4; ++u) { f.h[8+2*u] = (__bf16)base[16+2*u];   f.h[9+2*u] = (__bf16)base[16+2*u+1]; }
  return f.v;
}

__device__ inline v8f wmma_bf16(v16bf a, v16bf b, v8f c) {
  return __builtin_amdgcn_wmma_f32_16x16x32_bf16(false, a, false, b, (short)0, c, false, false);
}

// Fast, overflow-safe activations on native HW transcendentals (v_exp/v_log/v_rcp).
__device__ inline float rcp_(float x)  { return __builtin_amdgcn_rcpf(x); }
__device__ inline float sigm(float x)  { return rcp_(1.f + __expf(-x)); }
__device__ inline float eluf(float x)  { return x > 0.f ? x : __expf(x) - 1.f; }
__device__ inline float tanh_(float x) { return 1.f - 2.f * rcp_(__expf(2.f * x) + 1.f); }
__device__ inline float softplusf(float x) { return x > 20.f ? x : __logf(1.f + __expf(x)); }

__global__ __launch_bounds__(256, 1) void rssm_scan(RssmParams P) {
  __shared__ float  s_deter[2][16][D_];
  __shared__ float  s_stoch[2][16][32];
  __shared__ float  s_stat[16][2];
  __shared__ __bf16 s_bf[SBF_SIZE];

  const int tid  = threadIdx.x;
  const int lane = tid & 31;
  const int wv   = tid >> 5;
  const int hi   = lane >> 4;
  const int b0   = blockIdx.x * 16;

  for (int i = tid; i < 2 * 16 * D_; i += 256) ((float*)s_deter)[i] = 0.f;
  for (int i = tid; i < 2 * 16 * 32; i += 256) ((float*)s_stoch)[i] = 0.f;
  __syncthreads();

  for (int t = 0; t < T_; ++t) {
    for (int br = 0; br < 2; ++br) {
      const __bf16* w_inp = br ? P.inpf_wt  : P.inp_wt;
      const float*  b_inp = br ? P.inpf_b   : P.inp_b;
      const __bf16* w_gru = br ? P.gruf_wt  : P.gru_wt;
      const float*  b_gru = br ? P.gruf_b   : P.gru_b;
      const float*  ls_g  = br ? P.gruf_ls  : P.gru_ls;
      const float*  lb_g  = br ? P.gruf_lb  : P.gru_lb;
      const __bf16* w_img = br ? P.imgof_wt : P.imgo_wt;
      const float*  b_img = br ? P.imgof_b  : P.imgo_b;
      const __bf16* w_obs = br ? P.obsof_wt : P.obso_wt;
      const float*  b_obs = br ? P.obsof_b  : P.obso_b;
      const __bf16* w_ims = br ? P.imsf_wt  : P.ims_wt;
      const float*  b_ims = br ? P.imsf_b   : P.ims_b;
      const __bf16* w_ost = br ? P.obsf_wt  : P.obs_wt;
      const float*  b_ost = br ? P.obsf_b   : P.obs_b;

      const float* esrc = P.embed + ((long long)b0 * T_ + t) * (2 * E_) + (br ? E_ : 0);
      const long long erow = (long long)T_ * (2 * E_);

      // ---- P0: GEMM1 input [stoch|action] (bf16, Kpad=64) + GRU tail [deter|0] ----
      for (int i = tid; i < 16 * 64; i += 256) {
        int r = i >> 6, k = i & 63;
        float v = 0.f;
        if (k < S_) v = s_stoch[br][r][k];
        else if (!br && k >= 32 && k < 32 + A_)
          v = P.action[((long long)(b0 + r) * T_ + t) * A_ + (k - 32)];
        s_bf[OFF_A + i] = (__bf16)v;
      }
      for (int i = tid; i < 16 * 216; i += 256) {
        int r = i / 216, c = i % 216;
        float v = (c < D_) ? s_deter[br][r][c] : 0.f;
        s_bf[OFF_XH + r * 416 + 200 + c] = (__bf16)v;
      }
      // Prefetch this step's embed half for the obs GEMM (16 rows x 4KB).
      for (int i = tid; i < 16 * 64; i += 256)
        __builtin_prefetch(esrc + (long long)(i >> 6) * erow + (i & 63) * 16, 0, 1);
      __syncthreads();

      // ---- P1: x = elu([stoch|a] @ inp_w + b) -> XH[:, :200] ----
      {
        const int kc_n = br ? 1 : 2;
        const int dk_i = br ? 32 : 64;
        for (int jt = wv; jt < 13; jt += 8) {
          int n0 = jt * 16;
          v8f acc = {};
          for (int kc = 0; kc < kc_n; ++kc) {
            v16bf a = ld_a_lds(s_bf + OFF_A, 64, kc * 32, lane);
            v16bf b = ld_b_gl(w_inp, dk_i, n0, kc * 32, lane);
            acc = wmma_bf16(a, b, acc);
          }
          int n = n0 + (lane & 15);
          if (n < H_) {
#pragma unroll
            for (int i = 0; i < 8; ++i) {
              int r = i + 8 * hi;
              s_bf[OFF_XH + r * 416 + n] = (__bf16)eluf(acc[i] + b_inp[n]);
            }
          }
        }
      }
      __syncthreads();

      // ---- P2: parts = [x|deter] @ gru_w + b  (pre-LN, bf16) ----
      // Software-pipelined K loop: issue chunk kc+1 loads before the kc WMMA.
      for (int jt = wv; jt < 38; jt += 8) {
        int n0 = jt * 16;
        v8f acc = {};
        v16bf a = ld_a_lds(s_bf + OFF_XH, 416, 0, lane);
        v16bf b = ld_b_gl(w_gru, 416, n0, 0, lane);
        for (int kc = 1; kc < 13; ++kc) {
          v16bf an = ld_a_lds(s_bf + OFF_XH, 416, kc * 32, lane);
          v16bf bn = ld_b_gl(w_gru, 416, n0, kc * 32, lane);
          acc = wmma_bf16(a, b, acc);
          a = an; b = bn;
        }
        acc = wmma_bf16(a, b, acc);
        int n = n0 + (lane & 15);
        if (n < 600) {
#pragma unroll
          for (int i = 0; i < 8; ++i) {
            int r = i + 8 * hi;
            s_bf[OFF_PARTS + r * 600 + n] = (__bf16)(acc[i] + b_gru[n]);
          }
        }
      }
      __syncthreads();

      // ---- P3: LayerNorm stats per row over 600 ----
      for (int r = wv; r < 16; r += 8) {
        float s = 0.f, s2 = 0.f;
        for (int c = lane; c < 600; c += 32) {
          float v = (float)s_bf[OFF_PARTS + r * 600 + c];
          s += v; s2 += v * v;
        }
        for (int o = 16; o >= 1; o >>= 1) { s += __shfl_xor(s, o, 32); s2 += __shfl_xor(s2, o, 32); }
        if (lane == 0) {
          float m = s * (1.f / 600.f);
          float var = s2 * (1.f / 600.f) - m * m;
          s_stat[r][0] = m;
          s_stat[r][1] = rsqrtf(var + 1e-5f);
        }
      }
      __syncthreads();

      // ---- P4: GRU gates + new deter (state, DB bf16, global out) ----
      for (int i = tid; i < 16 * D_; i += 256) {
        int r = i / D_, j = i % D_;
        float mean = s_stat[r][0], rstd = s_stat[r][1];
        const __bf16* pr = s_bf + OFF_PARTS + r * 600;
        float pre_r = ((float)pr[j]       - mean) * rstd * ls_g[j]       + lb_g[j];
        float pre_c = ((float)pr[200 + j] - mean) * rstd * ls_g[200 + j] + lb_g[200 + j];
        float pre_u = ((float)pr[400 + j] - mean) * rstd * ls_g[400 + j] + lb_g[400 + j];
        float reset = sigm(pre_r);
        float cand  = tanh_(reset * pre_c);
        float upd   = sigm(pre_u - 1.f);
        float nd = upd * cand + (1.f - upd) * s_deter[br][r][j];
        s_deter[br][r][j] = nd;
        s_bf[OFF_DB + r * 224 + j] = (__bf16)nd;
        float* o = P.out + ((long long)(b0 + r) * T_ + t) * OUTW;
        if (!br) { o[90 + j] = nd; o[670 + j] = nd; }
        else     { o[380 + j] = nd; o[960 + j] = nd; }
      }
      for (int i = tid; i < 16 * 24; i += 256) {  // zero K-pad of DB
        int r = i / 24, c = 200 + i % 24;
        s_bf[OFF_DB + r * 224 + c] = (__bf16)0.f;
      }
      __syncthreads();

      // ---- G2: xo = elu(deter @ imgo_w); xo2 = elu([deter|embed] @ obso_w) ----
      for (int i = tid; i < 16 * 24; i += 256) {  // zero K-pads of XO/XO2
        int r = i / 24, c = 200 + i % 24;
        s_bf[OFF_XO  + r * 224 + c] = (__bf16)0.f;
        s_bf[OFF_XO2 + r * 224 + c] = (__bf16)0.f;
      }
      for (int jt = wv; jt < 13; jt += 8) {       // imgo (pipelined)
        int n0 = jt * 16;
        v8f acc = {};
        v16bf a = ld_a_lds(s_bf + OFF_DB, 224, 0, lane);
        v16bf b = ld_b_gl(w_img, 224, n0, 0, lane);
        for (int kc = 1; kc < 7; ++kc) {
          v16bf an = ld_a_lds(s_bf + OFF_DB, 224, kc * 32, lane);
          v16bf bn = ld_b_gl(w_img, 224, n0, kc * 32, lane);
          acc = wmma_bf16(a, b, acc);
          a = an; b = bn;
        }
        acc = wmma_bf16(a, b, acc);
        int n = n0 + (lane & 15);
        if (n < H_) {
#pragma unroll
          for (int i = 0; i < 8; ++i) {
            int r = i + 8 * hi;
            s_bf[OFF_XO + r * 224 + n] = (__bf16)eluf(acc[i] + b_img[n]);
          }
        }
      }
      {                                           // obso: dual-tile per wave, A reused, pipelined
        int t0 = wv, t1 = wv + 8;
        bool has1 = (t1 < 13);
        int n00 = t0 * 16;
        int n01 = has1 ? t1 * 16 : n00;           // clamped: always in-bounds
        v8f acc0 = {}, acc1 = {};
        v16bf a, c0, c1;
        a  = ld_a_lds(s_bf + OFF_DB, 224, 0, lane);
        c0 = ld_b_gl(w_obs, 1248, n00, 0, lane);
        c1 = has1 ? ld_b_gl(w_obs, 1248, n01, 0, lane) : c0;
        for (int kc = 1; kc < 7; ++kc) {          // deter part (K rows 0..223, pad*W=0)
          v16bf an  = ld_a_lds(s_bf + OFF_DB, 224, kc * 32, lane);
          v16bf bn0 = ld_b_gl(w_obs, 1248, n00, kc * 32, lane);
          v16bf bn1 = has1 ? ld_b_gl(w_obs, 1248, n01, kc * 32, lane) : bn0;
          acc0 = wmma_bf16(a, c0, acc0);
          if (has1) acc1 = wmma_bf16(a, c1, acc1);
          a = an; c0 = bn0; c1 = bn1;
        }
        {                                         // phase handoff to embed part
          v16bf an  = ld_a_glf32(esrc, erow, 0, lane);
          v16bf bn0 = ld_b_gl(w_obs, 1248, n00, 200, lane);
          v16bf bn1 = has1 ? ld_b_gl(w_obs, 1248, n01, 200, lane) : bn0;
          acc0 = wmma_bf16(a, c0, acc0);
          if (has1) acc1 = wmma_bf16(a, c1, acc1);
          a = an; c0 = bn0; c1 = bn1;
        }
        for (int kc = 1; kc < 32; ++kc) {         // embed part (weight rows 200..1223)
          v16bf an  = ld_a_glf32(esrc, erow, kc * 32, lane);
          v16bf bn0 = ld_b_gl(w_obs, 1248, n00, 200 + kc * 32, lane);
          v16bf bn1 = has1 ? ld_b_gl(w_obs, 1248, n01, 200 + kc * 32, lane) : bn0;
          acc0 = wmma_bf16(a, c0, acc0);
          if (has1) acc1 = wmma_bf16(a, c1, acc1);
          a = an; c0 = bn0; c1 = bn1;
        }
        acc0 = wmma_bf16(a, c0, acc0);
        if (has1) acc1 = wmma_bf16(a, c1, acc1);

        int n = n00 + (lane & 15);
        if (n < H_) {
#pragma unroll
          for (int i = 0; i < 8; ++i) {
            int r = i + 8 * hi;
            s_bf[OFF_XO2 + r * 224 + n] = (__bf16)eluf(acc0[i] + b_obs[n]);
          }
        }
        if (has1) {
          int n1 = n01 + (lane & 15);
          if (n1 < H_) {
#pragma unroll
            for (int i = 0; i < 8; ++i) {
              int r = i + 8 * hi;
              s_bf[OFF_XO2 + r * 224 + n1] = (__bf16)eluf(acc1[i] + b_obs[n1]);
            }
          }
        }
      }
      __syncthreads();

      // ---- G3: prior stats (ims from XO) and posterior stats (obs from XO2) ----
      {
        bool isPrior = (wv < 4);
        int jt = isPrior ? wv : (wv - 4);
        int n0 = jt * 16;
        const __bf16* wS = isPrior ? w_ims : w_ost;
        const float*  bS = isPrior ? b_ims : b_ost;
        int offA = isPrior ? OFF_XO : OFF_XO2;
        v8f acc = {};
        v16bf a = ld_a_lds(s_bf + offA, 224, 0, lane);
        v16bf b = ld_b_gl(wS, 224, n0, 0, lane);
        for (int kc = 1; kc < 7; ++kc) {
          v16bf an = ld_a_lds(s_bf + offA, 224, kc * 32, lane);
          v16bf bn = ld_b_gl(wS, 224, n0, kc * 32, lane);
          acc = wmma_bf16(a, b, acc);
          a = an; b = bn;
        }
        acc = wmma_bf16(a, b, acc);
        int n = n0 + (lane & 15);
        if (n < 2 * S_) {
#pragma unroll
          for (int i = 0; i < 8; ++i) {
            int r = i + 8 * hi;
            float v = acc[i] + bS[n];
            float* o = P.out + ((long long)(b0 + r) * T_ + t) * OUTW;
            if (n < S_) {
              if (isPrior) { o[(br ? 870 : 580) + n] = v; o[(br ? 930 : 640) + n] = v; }
              else {
                o[(br ? 290 : 0) + n] = v; o[(br ? 350 : 60) + n] = v;
                s_stoch[br][r][n] = v;   // new stoch = posterior mean (mode)
              }
            } else {
              float sp = softplusf(v) + 0.1f;
              o[(isPrior ? (br ? 900 : 610) : (br ? 320 : 30)) + (n - S_)] = sp;
            }
          }
        }
      }
      __syncthreads();
    }  // br
  }    // t
}

// Prologue: convert+transpose weight [din,dout] f32 -> WT [dn][dk] bf16, zero-padded.
// remap=1 (inp_w): dst k 0..29 <- rows 0..29 (stoch), k 30..31 <- 0, k 32..37 <- rows 30..35 (action).
__global__ void prep_weight(const float* __restrict__ src, __bf16* __restrict__ dst,
                            int din, int dout, int dk, int dn, int remap) {
  int total = dk * dn;
  for (int idx = blockIdx.x * 256 + threadIdx.x; idx < total; idx += gridDim.x * 256) {
    int n = idx / dk, k = idx - n * dk;
    float v = 0.f;
    if (n < dout) {
      if (remap) {
        if (k < 30)                 v = src[k * dout + n];
        else if (k >= 32 && k < 38) v = src[(k - 2) * dout + n];
      } else if (k < din) {
        v = src[k * dout + n];
      }
    }
    dst[(size_t)n * dk + k] = (__bf16)v;
  }
}

extern "C" void kernel_launch(void* const* d_in, const int* in_sizes, int n_in,
                              void* d_out, int out_size, void* d_ws, size_t ws_size,
                              hipStream_t stream) {
  (void)in_sizes; (void)n_in; (void)out_size; (void)ws_size;
  auto F = [&](int i) { return (const float*)d_in[i]; };
  __bf16* ws = (__bf16*)d_ws;

  // d_in order: 0 embed, 1 action, 2 inp_w, 3 inp_b, 4 inpf_w, 5 inpf_b,
  // 6..9 gru_{w,b,ls,lb}, 10..13 gruf_{w,b,ls,lb}, 14 imgo_w, 15 imgo_b,
  // 16 imgof_w, 17 imgof_b, 18 obso_w, 19 obso_b, 20 obsof_w, 21 obsof_b,
  // 22 ims_w, 23 ims_b, 24 imsf_w, 25 imsf_b, 26 obs_w, 27 obs_b, 28 obsf_w, 29 obsf_b
  struct WDesc { int idx, din, dout, dk, dn, remap; };
  const WDesc wd[12] = {
    { 2,   36, 200,   64, 208, 1},  // inp
    { 4,   30, 200,   32, 208, 0},  // inpf
    { 6,  400, 600,  416, 608, 0},  // gru
    {10,  400, 600,  416, 608, 0},  // gruf
    {14,  200, 200,  224, 208, 0},  // imgo
    {16,  200, 200,  224, 208, 0},  // imgof
    {18, 1224, 200, 1248, 208, 0},  // obso
    {20, 1224, 200, 1248, 208, 0},  // obsof
    {22,  200,  60,  224,  64, 0},  // ims
    {24,  200,  60,  224,  64, 0},  // imsf
    {26,  200,  60,  224,  64, 0},  // obs
    {28,  200,  60,  224,  64, 0},  // obsf
  };
  __bf16* wt[12];
  size_t off = 0;
  for (int i = 0; i < 12; ++i) {
    wt[i] = ws + off;
    int total = wd[i].dk * wd[i].dn;
    int blocks = (total + 255) / 256;
    if (blocks > 1024) blocks = 1024;
    prep_weight<<<blocks, 256, 0, stream>>>(F(wd[i].idx), wt[i],
        wd[i].din, wd[i].dout, wd[i].dk, wd[i].dn, wd[i].remap);
    off += (size_t)total;
  }

  RssmParams P;
  P.embed  = F(0);  P.action = F(1);
  P.inp_b  = F(3);  P.inpf_b = F(5);
  P.gru_b  = F(7);  P.gru_ls = F(8);  P.gru_lb = F(9);
  P.gruf_b = F(11); P.gruf_ls = F(12); P.gruf_lb = F(13);
  P.imgo_b = F(15); P.imgof_b = F(17);
  P.obso_b = F(19); P.obsof_b = F(21);
  P.ims_b  = F(23); P.imsf_b = F(25);
  P.obs_b  = F(27); P.obsf_b = F(29);
  P.inp_wt  = wt[0]; P.inpf_wt  = wt[1]; P.gru_wt  = wt[2]; P.gruf_wt  = wt[3];
  P.imgo_wt = wt[4]; P.imgof_wt = wt[5]; P.obso_wt = wt[6]; P.obsof_wt = wt[7];
  P.ims_wt  = wt[8]; P.imsf_wt  = wt[9]; P.obs_wt  = wt[10]; P.obsf_wt = wt[11];
  P.out = (float*)d_out;

  rssm_scan<<<B_ / 16, 256, 0, stream>>>(P);
}
